// MultiHeadAttention_78168404787424
// MI455X (gfx1250) — matrix operand order repaired and checked by reading, not compiled, not run on hardware
//
#include <hip/hip_runtime.h>
#include <hip/hip_bf16.h>

typedef __attribute__((ext_vector_type(16))) _Float16 v16h;
typedef __attribute__((ext_vector_type(8)))  _Float16 v8h;
typedef __attribute__((ext_vector_type(8)))  float    v8f;
typedef __attribute__((ext_vector_type(4))) unsigned int u32x4;
typedef __attribute__((ext_vector_type(8))) int          i32x8;
typedef __attribute__((ext_vector_type(4))) int          i32x4;

#if __has_builtin(__builtin_amdgcn_tensor_load_to_lds) && \
    __has_builtin(__builtin_amdgcn_s_wait_tensorcnt)
#define USE_TDM 1
#else
#define USE_TDM 0
#endif

namespace {
constexpr int   kB     = 4;
constexpr int   kSeq   = 2048;
constexpr int   kDim   = 1024;
constexpr int   kH     = 16;
constexpr int   kHd    = 64;
constexpr int   kRows  = kB * kSeq;          // 8192
constexpr float kScale = 0.125f;             // 1/sqrt(64)
constexpr float kLog2e = 1.4426950408889634f;
}

__device__ __forceinline__ v16h make_v16(v8h lo, v8h hi) {
  v16h r;
#pragma unroll
  for (int i = 0; i < 8; ++i) { r[i] = lo[i]; r[i + 8] = hi[i]; }
  return r;
}

// A-operand fragment: 16 rows x 32 K, row-major [row, k] source.
__device__ __forceinline__ v16h load_fragA(const _Float16* __restrict__ p, int ld,
                                           int r0, int c0, int lane) {
  const _Float16* q = p + (size_t)(r0 + (lane & 15)) * ld + c0 + ((lane >> 4) << 3);
  v8h lo = *reinterpret_cast<const v8h*>(q);
  v8h hi = *reinterpret_cast<const v8h*>(q + 16);
  return make_v16(lo, hi);
}

// B-operand fragment: 16 cols x 32 K, from row-major B^T [n, k] source.
__device__ __forceinline__ v16h load_fragB(const _Float16* __restrict__ p, int ld,
                                           int n0, int c0, int lane) {
  const _Float16* q = p + (size_t)(n0 + (lane & 15)) * ld + c0 + ((lane >> 4) << 3);
  v8h lo = *reinterpret_cast<const v8h*>(q);
  v8h hi = *reinterpret_cast<const v8h*>(q + 16);
  return make_v16(lo, hi);
}

__device__ __forceinline__ v8f wmma16(v16h a, v16h b, v8f c) {
  return __builtin_amdgcn_wmma_f32_16x16x32_f16(false, a, false, b, (short)0, c,
                                                false, false);
}

#if USE_TDM
// TDM: DMA a 2D tile (rows x rowlen f16 elements, row stride `stride` elems)
// from global into LDS at byte offset lds_off. D# packing per CDNA5 ISA 8.3/8.4.
__device__ __forceinline__ void tdm_load_2d(const _Float16* gptr, unsigned lds_off,
                                            int rows, int rowlen, int stride) {
  unsigned long long ga = (unsigned long long)(uintptr_t)gptr;
  u32x4 g0;
  g0[0] = 1u;                                    // count=1, user mode, no gather
  g0[1] = lds_off;                               // lds_addr (bytes)
  g0[2] = (unsigned)ga;                          // global_addr[31:0]
  g0[3] = (unsigned)(ga >> 32) | 0x80000000u;    // global_addr[56:32] | type=2
  i32x8 g1;
  g1[0] = 1 << 16;                               // data_size = 2 bytes
  g1[1] = 0;                                     // no atomic barrier; dim0 lo16=0
  g1[2] = 0x10;                                  // tensor_dim0 = 1<<20 (no clip)
  g1[3] = 0x10 | (rowlen << 16);                 // tensor_dim1 = 1<<20; tile_dim0
  g1[4] = rows;                                  // tile_dim1 (tile_dim2 = 0)
  g1[5] = stride;                                // tensor_dim0_stride lo32
  g1[6] = 0;
  g1[7] = 0;
  i32x4 z4 = {0, 0, 0, 0};
  i32x8 z8 = {0, 0, 0, 0, 0, 0, 0, 0};
  __builtin_amdgcn_tensor_load_to_lds(g0, g1, z4, z4, z8, 0);
}
#endif

// ---------------------------------------------------------------- convert
__global__ __launch_bounds__(256) void cvt_f32_f16_kernel(
    const float* __restrict__ in, _Float16* __restrict__ out, int n) {
  int i = blockIdx.x * 256 + threadIdx.x;
  if (i < n) out[i] = (_Float16)in[i];
}

// ---------------------------------------------------------------- GEMM  Y = A @ W^T + bias
// 32x64 register tile per wave, software-pipelined k-loop (8 WMMA per k-step).
// mode 0: f16 row-major (Q,K). mode 1: f16 transposed to [B,H,hd,N] (V^T). mode 2: f32.
__global__ __launch_bounds__(256) void gemm_xwT_kernel(
    const _Float16* __restrict__ A, const _Float16* __restrict__ W,
    const float* __restrict__ bias, _Float16* __restrict__ Yh,
    float* __restrict__ Yf, int mode) {
  int lane = threadIdx.x & 31;
  int wave = threadIdx.x >> 5;
  int row0 = blockIdx.x * 32;
  int col0 = (blockIdx.y * 8 + wave) * 64;

  v8f acc[2][4];
#pragma unroll
  for (int t = 0; t < 4; ++t) {
    float bv = bias[col0 + 16 * t + (lane & 15)];
#pragma unroll
    for (int r = 0; r < 8; ++r) { acc[0][t][r] = bv; acc[1][t][r] = bv; }
  }

  v16h a0 = load_fragA(A, kDim, row0, 0, lane);
  v16h a1 = load_fragA(A, kDim, row0 + 16, 0, lane);
  v16h b[4];
#pragma unroll
  for (int t = 0; t < 4; ++t) b[t] = load_fragB(W, kDim, col0 + 16 * t, 0, lane);

  for (int c0 = 0; c0 < kDim; c0 += 32) {
    int cn = c0 + 32;
    v16h a0n = a0, a1n = a1, bn[4];
#pragma unroll
    for (int t = 0; t < 4; ++t) bn[t] = b[t];
    if (cn < kDim) {
      __builtin_prefetch(A + (size_t)(row0 + (lane & 15)) * kDim + cn + 32);
      a0n = load_fragA(A, kDim, row0, cn, lane);
      a1n = load_fragA(A, kDim, row0 + 16, cn, lane);
#pragma unroll
      for (int t = 0; t < 4; ++t) bn[t] = load_fragB(W, kDim, col0 + 16 * t, cn, lane);
    }
#pragma unroll
    for (int t = 0; t < 4; ++t) acc[0][t] = wmma16(a0, b[t], acc[0][t]);
#pragma unroll
    for (int t = 0; t < 4; ++t) acc[1][t] = wmma16(a1, b[t], acc[1][t]);
    a0 = a0n; a1 = a1n;
#pragma unroll
    for (int t = 0; t < 4; ++t) b[t] = bn[t];
  }

  int rsel = lane >> 4, csel = lane & 15;
#pragma unroll
  for (int rt = 0; rt < 2; ++rt) {
#pragma unroll
    for (int t = 0; t < 4; ++t) {
#pragma unroll
      for (int r = 0; r < 8; ++r) {
        int row = row0 + 16 * rt + r + 8 * rsel;
        int col = col0 + 16 * t + csel;
        float v = acc[rt][t][r];
        if (mode == 0) {
          Yh[(size_t)row * kDim + col] = (_Float16)v;
        } else if (mode == 1) {
          int bb = row >> 11, n = row & (kSeq - 1);
          int h = col >> 6, hd = col & (kHd - 1);
          Yh[((size_t)((bb * kH + h) * kHd + hd)) * kSeq + n] = (_Float16)v;
        } else {
          Yf[(size_t)row * kDim + col] = v;
        }
      }
    }
  }
}

// ---------------------------------------------------------------- flash attention
// One wave per 16-query tile per (b,h); 32-key chunks. K/V tiles staged into LDS
// by the Tensor Data Mover (double-buffered, s_wait_tensorcnt) when available.
__global__ __launch_bounds__(32) void attn_kernel(
    const _Float16* __restrict__ Q, const _Float16* __restrict__ K,
    const _Float16* __restrict__ Vt, _Float16* __restrict__ O) {
#if USE_TDM
  __shared__ __align__(16) _Float16 ldsK[2][32 * 64];  // key chunk, row-major [n, hd]
  __shared__ __align__(16) _Float16 ldsV[2][64 * 32];  // V^T chunk, row-major [hd, n]
#endif
  __shared__ __align__(16) _Float16 lds_p[16 * 40];    // P tile relayout scratch

  int lane = threadIdx.x & 31;
  int m0 = blockIdx.x * 16;
  int h  = blockIdx.y;
  int b  = blockIdx.z;

  const _Float16* Qp = Q + (size_t)b * kSeq * kDim + h * kHd;    // ld = kDim
  const _Float16* Kp = K + (size_t)b * kSeq * kDim + h * kHd;    // ld = kDim
  const _Float16* Vp = Vt + ((size_t)(b * kH + h) * kHd) * kSeq; // ld = kSeq

  v16h qa0 = load_fragA(Qp, kDim, m0, 0, lane);
  v16h qa1 = load_fragA(Qp, kDim, m0, 32, lane);

  v8f o[4] = {v8f{}, v8f{}, v8f{}, v8f{}};
  float mrow[8], ssum[8];
#pragma unroll
  for (int r = 0; r < 8; ++r) { mrow[r] = 0.f; ssum[r] = 0.f; }

#if USE_TDM
  // prefetch chunk 0
  tdm_load_2d(Kp, (unsigned)(uintptr_t)&ldsK[0][0], 32, 64, kDim);
  tdm_load_2d(Vp, (unsigned)(uintptr_t)&ldsV[0][0], 64, 32, kSeq);
#endif

  for (int it = 0; it < kSeq / 32; ++it) {
    int n0 = it * 32;
#if USE_TDM
    int cur = it & 1;
    if (it + 1 < kSeq / 32) {
      tdm_load_2d(Kp + (size_t)(n0 + 32) * kDim,
                  (unsigned)(uintptr_t)&ldsK[cur ^ 1][0], 32, 64, kDim);
      tdm_load_2d(Vp + (n0 + 32),
                  (unsigned)(uintptr_t)&ldsV[cur ^ 1][0], 64, 32, kSeq);
      __builtin_amdgcn_s_wait_tensorcnt(2);  // chunk `it` done (in-order per wave)
    } else {
      __builtin_amdgcn_s_wait_tensorcnt(0);
    }
    const _Float16* kt = &ldsK[cur][0];  // ld 64
    const _Float16* vt = &ldsV[cur][0];  // ld 32
    v16h kb00 = load_fragB(kt, 64, 0, 0, lane);
    v16h kb01 = load_fragB(kt, 64, 0, 32, lane);
    v16h kb10 = load_fragB(kt, 64, 16, 0, lane);
    v16h kb11 = load_fragB(kt, 64, 16, 32, lane);
#else
    v16h kb00 = load_fragB(Kp, kDim, n0, 0, lane);
    v16h kb01 = load_fragB(Kp, kDim, n0, 32, lane);
    v16h kb10 = load_fragB(Kp, kDim, n0 + 16, 0, lane);
    v16h kb11 = load_fragB(Kp, kDim, n0 + 16, 32, lane);
#endif
    v8f s0 = {}, s1 = {};
    s0 = wmma16(qa0, kb00, s0);
    s0 = wmma16(qa1, kb01, s0);
    s1 = wmma16(qa0, kb10, s1);
    s1 = wmma16(qa1, kb11, s1);

    float a0[8], a1[8], cm[8];
#pragma unroll
    for (int r = 0; r < 8; ++r) {
      s0[r] *= kScale;  s1[r] *= kScale;
      a0[r] = fabsf(s0[r]);  a1[r] = fabsf(s1[r]);
      cm[r] = fmaxf(a0[r], a1[r]);
    }
#pragma unroll
    for (int ofs = 1; ofs < 16; ofs <<= 1)
#pragma unroll
      for (int r = 0; r < 8; ++r) cm[r] = fmaxf(cm[r], __shfl_xor(cm[r], ofs, 32));

    float e0[8], e1[8], rs[8], fac[8];
#pragma unroll
    for (int r = 0; r < 8; ++r) {
      float mn = fmaxf(mrow[r], cm[r]);
      fac[r] = exp2f((mrow[r] - mn) * kLog2e);
      mrow[r] = mn;
      e0[r] = exp2f((a0[r] - mn) * kLog2e);
      e1[r] = exp2f((a1[r] - mn) * kLog2e);
      rs[r] = e0[r] + e1[r];
    }
#pragma unroll
    for (int ofs = 1; ofs < 16; ofs <<= 1)
#pragma unroll
      for (int r = 0; r < 8; ++r) rs[r] += __shfl_xor(rs[r], ofs, 32);

#pragma unroll
    for (int r = 0; r < 8; ++r) {
      ssum[r] = ssum[r] * fac[r] + rs[r];
#pragma unroll
      for (int j = 0; j < 4; ++j) o[j][r] *= fac[r];
    }

    // P = sign(S)*exp(|S|-m) -> LDS (row-major) -> reload as A fragment
    {
      int prow = 8 * (lane >> 4);
      int pcol = lane & 15;
#pragma unroll
      for (int r = 0; r < 8; ++r) {
        float p0 = s0[r] > 0.f ? e0[r] : (s0[r] < 0.f ? -e0[r] : 0.f);
        float p1 = s1[r] > 0.f ? e1[r] : (s1[r] < 0.f ? -e1[r] : 0.f);
        lds_p[(prow + r) * 40 + pcol] = (_Float16)p0;
        lds_p[(prow + r) * 40 + 16 + pcol] = (_Float16)p1;
      }
    }
    v16h pa;
    {
      const _Float16* q = &lds_p[(lane & 15) * 40 + ((lane >> 4) << 3)];
      v8h lo = *reinterpret_cast<const v8h*>(q);
      v8h hi = *reinterpret_cast<const v8h*>(q + 16);
      pa = make_v16(lo, hi);
    }

#pragma unroll
    for (int j = 0; j < 4; ++j) {
#if USE_TDM
      v16h vb = load_fragB(vt, 32, j * 16, 0, lane);
#else
      v16h vb = load_fragB(Vp, kSeq, j * 16, n0, lane);
#endif
      o[j] = wmma16(pa, vb, o[j]);
    }
  }

  int rsel = lane >> 4, csel = lane & 15;
#pragma unroll
  for (int j = 0; j < 4; ++j) {
#pragma unroll
    for (int r = 0; r < 8; ++r) {
      size_t row = (size_t)b * kSeq + m0 + r + 8 * rsel;
      int col = h * kHd + j * 16 + csel;
      O[row * kDim + col] = (_Float16)(o[j][r] / ssum[r]);
    }
  }
}

// ---------------------------------------------------------------- launcher
extern "C" void kernel_launch(void* const* d_in, const int* in_sizes, int n_in,
                              void* d_out, int out_size, void* d_ws, size_t ws_size,
                              hipStream_t stream) {
  (void)in_sizes; (void)n_in; (void)out_size; (void)ws_size;
  const float* x  = (const float*)d_in[0];
  const float* Wq = (const float*)d_in[1];
  const float* bq = (const float*)d_in[2];
  const float* Wk = (const float*)d_in[3];
  const float* bk = (const float*)d_in[4];
  const float* Wv = (const float*)d_in[5];
  const float* bv = (const float*)d_in[6];
  const float* Wo = (const float*)d_in[7];
  const float* bo = (const float*)d_in[8];
  float* out = (float*)d_out;

  const size_t XE = (size_t)kRows * kDim;  // 8 M halves
  const size_t WE = (size_t)kDim * kDim;   // 1 M halves
  _Float16* x16  = (_Float16*)d_ws;
  _Float16* Wq16 = x16 + XE;
  _Float16* Wk16 = Wq16 + WE;
  _Float16* Wv16 = Wk16 + WE;
  _Float16* Wo16 = Wv16 + WE;
  _Float16* Q16  = Wo16 + WE;
  _Float16* K16  = Q16 + XE;
  _Float16* Vt16 = K16 + XE;
  _Float16* A16  = Vt16 + XE;  // total ~88 MB

  cvt_f32_f16_kernel<<<(int)(XE / 256), 256, 0, stream>>>(x, x16, (int)XE);
  cvt_f32_f16_kernel<<<(int)(WE / 256), 256, 0, stream>>>(Wq, Wq16, (int)WE);
  cvt_f32_f16_kernel<<<(int)(WE / 256), 256, 0, stream>>>(Wk, Wk16, (int)WE);
  cvt_f32_f16_kernel<<<(int)(WE / 256), 256, 0, stream>>>(Wv, Wv16, (int)WE);
  cvt_f32_f16_kernel<<<(int)(WE / 256), 256, 0, stream>>>(Wo, Wo16, (int)WE);

  dim3 gg(kRows / 32, kDim / 512), gb(256);
  gemm_xwT_kernel<<<gg, gb, 0, stream>>>(x16, Wq16, bq, Q16, nullptr, 0);
  gemm_xwT_kernel<<<gg, gb, 0, stream>>>(x16, Wk16, bk, K16, nullptr, 0);
  gemm_xwT_kernel<<<gg, gb, 0, stream>>>(x16, Wv16, bv, Vt16, nullptr, 1);

  attn_kernel<<<dim3(kSeq / 16, kH, kB), 32, 0, stream>>>(Q16, K16, Vt16, A16);

  gemm_xwT_kernel<<<gg, gb, 0, stream>>>(A16, Wo16, bo, nullptr, out, 2);
}
